// GNNModel_22789096472974
// MI455X (gfx1250) — matrix operand order repaired and checked by reading, not compiled, not run on hardware
//
#include <hip/hip_runtime.h>

typedef __bf16 bf16;
typedef __attribute__((ext_vector_type(16))) bf16  v16bf;
typedef __attribute__((ext_vector_type(8)))  bf16  v8bf;
typedef __attribute__((ext_vector_type(4)))  bf16  v4bf;
typedef __attribute__((ext_vector_type(8)))  float v8f;

typedef __attribute__((address_space(3))) char lds_char_t;

#define N_NODES 100000
#define E_EDGES 1600000
#define FDIM    128
#define M_TILE  32
#define NEG_SLOPE 0.01f
#define KEEP_SCALE (1.0f / 0.9f)
#define DROP_THRESH 429496730u   // P(drop)=0.1 on 32-bit hash

// ---------------------------------------------------------------- utilities
__global__ void k_zero4(float4* __restrict__ p, int n4) {
    int i = blockIdx.x * blockDim.x + threadIdx.x;
    if (i < n4) p[i] = make_float4(0.f, 0.f, 0.f, 0.f);
}

__global__ void k_degree(const int* __restrict__ dst, float* __restrict__ deg, int e) {
    int i = blockIdx.x * blockDim.x + threadIdx.x;
    if (i < e) atomicAdd(&deg[dst[i]], 1.0f);
}

__global__ void k_rsqrt_inplace(float* __restrict__ d, int n) {
    int i = blockIdx.x * blockDim.x + threadIdx.x;
    if (i < n) d[i] = rsqrtf(d[i] + 1.0f);   // +1 for self loop
}

// W [k][n] f32  ->  Wt [n][k] bf16  (fragment-friendly layout)
__global__ void k_w_to_bf16_T(const float* __restrict__ W, bf16* __restrict__ Wt) {
    int i = blockIdx.x * blockDim.x + threadIdx.x;   // < 128*128
    int k = i >> 7, n = i & 127;
    Wt[n * FDIM + k] = (bf16)W[i];
}

// ---------------------------------------------------------------- WMMA GEMM
// H[m0:m0+32, :] = X[m0:m0+32, :] (f32) @ W (via Wt bf16 [n][k]), output bf16.
// 256 threads = 8 waves; wave w owns columns [16w,16w+16) x two 16-row strips.
__global__ __launch_bounds__(256) void k_gemm_wmma(const float* __restrict__ X,
                                                   const bf16*  __restrict__ Wt,
                                                   bf16*        __restrict__ Hout) {
    __shared__ bf16 sWt[FDIM * FDIM];    // 32 KB, [n][k]
    __shared__ bf16 sA[M_TILE * FDIM];   //  8 KB, [m][k]

    const int tid = threadIdx.x;
    const int m0  = blockIdx.x * M_TILE;

    // ---- stage Wt (straight byte copy) via CDNA5 async global->LDS copy
    {
        const char* g = (const char*)Wt;
        char* l = (char*)&sWt[0];
        #pragma unroll
        for (int off = 0; off < FDIM * FDIM * 2; off += 256 * 16) {
            lds_char_t* laddr = (lds_char_t*)(l + off + tid * 16);
            const char*  gaddr = g + off + tid * 16;
            asm volatile("global_load_async_to_lds_b128 %0, %1, off"
                         :: "v"(laddr), "v"(gaddr) : "memory");
        }
    }
    // ---- stage A strip (f32 -> bf16 conversion, regular DS stores)
    {
        const float* xrow = X + (size_t)m0 * FDIM;
        #pragma unroll
        for (int i = tid; i < M_TILE * FDIM; i += 256) sA[i] = (bf16)xrow[i];
    }
    asm volatile("s_wait_asynccnt 0" ::: "memory");
    __syncthreads();

    const int wave = tid >> 5;
    const int lane = tid & 31;
    const int n0   = wave * 16;
    const int mrow = lane & 15;     // A row within strip
    const int half = lane >> 4;     // lane group 0/1
    const int ncol = lane & 15;     // B/D column

    v8f acc0 = {};
    v8f acc1 = {};
#pragma unroll
    for (int k0 = 0; k0 < FDIM; k0 += 32) {
        // B 32x16 bf16: lane column = ncol, K = k0 + half*16 + i  (contiguous 16)
        v16bf b;
        {
            const bf16* bp = &sWt[(n0 + ncol) * FDIM + k0 + half * 16];
            ((v8bf*)&b)[0] = *(const v8bf*)(bp);
            ((v8bf*)&b)[1] = *(const v8bf*)(bp + 8);
        }
        // A 16x32 bf16 per strip: K = k0 + half*16 + i  (contiguous 16)
        v16bf a0, a1;
        {
            const bf16* ap0 = &sA[(mrow)      * FDIM + k0 + half * 16];
            const bf16* ap1 = &sA[(mrow + 16) * FDIM + k0 + half * 16];
            ((v8bf*)&a0)[0] = *(const v8bf*)(ap0);
            ((v8bf*)&a0)[1] = *(const v8bf*)(ap0 + 8);
            ((v8bf*)&a1)[0] = *(const v8bf*)(ap1);
            ((v8bf*)&a1)[1] = *(const v8bf*)(ap1 + 8);
        }
        acc0 = __builtin_amdgcn_wmma_f32_16x16x32_bf16(false, a0, false, b,
                                                       (short)0, acc0, false, false);
        acc1 = __builtin_amdgcn_wmma_f32_16x16x32_bf16(false, a1, false, b,
                                                       (short)0, acc1, false, false);
    }

    // D 16x16 f32: VGPR r -> M = half*8 + r, N = ncol
    bf16* o0 = Hout + ((size_t)(m0 + half * 8))      * FDIM + n0 + ncol;
    bf16* o1 = Hout + ((size_t)(m0 + 16 + half * 8)) * FDIM + n0 + ncol;
#pragma unroll
    for (int r = 0; r < 8; ++r) {
        o0[(size_t)r * FDIM] = (bf16)acc0[r];
        o1[(size_t)r * FDIM] = (bf16)acc1[r];
    }
}

// ---------------------------------------------------------------- scatter
// One wave per edge; each lane handles 4 channels (8B bf16 load, 4 f32 atomics).
__global__ __launch_bounds__(256) void k_scatter(const int*  __restrict__ ei,
                                                 const bf16* __restrict__ H,
                                                 const float* __restrict__ dinv,
                                                 float*      __restrict__ agg) {
    const int gw = (blockIdx.x * 256 + threadIdx.x) >> 5;   // edge id
    if (gw >= E_EDGES) return;
    const int lane = threadIdx.x & 31;
    const int src  = ei[gw];
    const int dst  = ei[E_EDGES + gw];
    const float norm = dinv[src] * dinv[dst];

    const v4bf hv = *(const v4bf*)(H + (size_t)src * FDIM + lane * 4);
    float* ap = agg + (size_t)dst * FDIM + lane * 4;
    atomicAdd(ap + 0, (float)hv[0] * norm);
    atomicAdd(ap + 1, (float)hv[1] * norm);
    atomicAdd(ap + 2, (float)hv[2] * norm);
    atomicAdd(ap + 3, (float)hv[3] * norm);
}

// ------------------------------------------------- self-loop + bias + act + dropout
__global__ __launch_bounds__(256) void k_post(float* __restrict__ agg,
                                              const bf16* __restrict__ H,
                                              const float* __restrict__ dinv,
                                              const float* __restrict__ bias,
                                              unsigned seed) {
    const unsigned i = blockIdx.x * 256 + threadIdx.x;    // < N*FDIM
    const int node = i >> 7;
    const int c    = i & 127;
    const float di = dinv[node];
    float v = agg[i] + (float)H[i] * di * di + bias[c];
    v = v > 0.f ? v : NEG_SLOPE * v;
    unsigned x = i * 2654435761u + seed;
    x ^= x >> 16; x *= 0x7feb352du;
    x ^= x >> 15; x *= 0x846ca68bu;
    x ^= x >> 16;
    agg[i] = (x >= DROP_THRESH) ? v * KEEP_SCALE : 0.0f;
}

// ---------------------------------------------------------------- final linear
__global__ __launch_bounds__(256) void k_final(const float* __restrict__ H,
                                               const float* __restrict__ Wl,
                                               const float* __restrict__ bl,
                                               float* __restrict__ out) {
    const int n = blockIdx.x * blockDim.x + threadIdx.x;
    if (n >= N_NODES) return;
    const float4* row = (const float4*)(H + (size_t)n * FDIM);
    const float4* w4  = (const float4*)Wl;
    float s = 0.f;
#pragma unroll
    for (int i = 0; i < FDIM / 4; ++i) {
        float4 v = row[i];
        float4 w = w4[i];
        s += v.x * w.x + v.y * w.y + v.z * w.z + v.w * w.w;
    }
    out[n] = s + bl[0];
}

// ---------------------------------------------------------------- launch
extern "C" void kernel_launch(void* const* d_in, const int* in_sizes, int n_in,
                              void* d_out, int out_size, void* d_ws, size_t ws_size,
                              hipStream_t stream) {
    const float* x  = (const float*)d_in[0];
    const int*   ei = (const int*)  d_in[1];   // [2, E]
    const float* W1 = (const float*)d_in[2];
    const float* b1 = (const float*)d_in[3];
    const float* W2 = (const float*)d_in[4];
    const float* b2 = (const float*)d_in[5];
    const float* Wl = (const float*)d_in[6];
    const float* bl = (const float*)d_in[7];
    float* out = (float*)d_out;

    // workspace layout (all offsets 16B aligned)
    char* ws = (char*)d_ws;
    float* dinv = (float*)ws;                                      // N f32 (400000 B)
    bf16*  Wt1  = (bf16*)(ws + 400000);                            // 128x128 bf16 [n][k]
    bf16*  Wt2  = (bf16*)(ws + 400000 + 32768);                    // 128x128 bf16 [n][k]
    bf16*  hb   = (bf16*)(ws + 400000 + 65536);                    // N*128 bf16 (25.6 MB)
    float* agg  = (float*)(ws + 400000 + 65536 + (size_t)N_NODES * FDIM * 2); // N*128 f32

    const int NF = N_NODES * FDIM;          // 12,800,000

    // 1) degrees -> dinv_sqrt
    k_zero4<<<(N_NODES / 4 + 255) / 256, 256, 0, stream>>>((float4*)dinv, N_NODES / 4);
    k_degree<<<E_EDGES / 256, 256, 0, stream>>>(ei + E_EDGES, dinv, E_EDGES);
    k_rsqrt_inplace<<<(N_NODES + 255) / 256, 256, 0, stream>>>(dinv, N_NODES);

    // 2) weights -> transposed bf16
    k_w_to_bf16_T<<<(FDIM * FDIM) / 256, 256, 0, stream>>>(W1, Wt1);
    k_w_to_bf16_T<<<(FDIM * FDIM) / 256, 256, 0, stream>>>(W2, Wt2);

    // ---- layer 1
    k_gemm_wmma<<<N_NODES / M_TILE, 256, 0, stream>>>(x, Wt1, hb);
    k_zero4<<<(NF / 4) / 256, 256, 0, stream>>>((float4*)agg, NF / 4);
    k_scatter<<<E_EDGES / 8, 256, 0, stream>>>(ei, hb, dinv, agg);
    k_post<<<NF / 256, 256, 0, stream>>>(agg, hb, dinv, b1, 0x9E3779B9u);

    // ---- layer 2 (agg holds layer-1 activations, f32)
    k_gemm_wmma<<<N_NODES / M_TILE, 256, 0, stream>>>(agg, Wt2, hb);
    k_zero4<<<(NF / 4) / 256, 256, 0, stream>>>((float4*)agg, NF / 4);
    k_scatter<<<E_EDGES / 8, 256, 0, stream>>>(ei, hb, dinv, agg);
    k_post<<<NF / 256, 256, 0, stream>>>(agg, hb, dinv, b2, 0x85EBCA6Bu);

    // ---- final projection
    k_final<<<(N_NODES + 255) / 256, 256, 0, stream>>>(agg, Wl, bl, out);
}
